// XLAttention_73014444032009
// MI455X (gfx1250) — matrix operand order repaired and checked by reading, not compiled, not run on hardware
//
#include <hip/hip_runtime.h>
#include <hip/hip_bf16.h>

// XL-attention for MI455X (gfx1250, wave32, WMMA), round 4.
//  - all matmuls bf16 / f32 accumulate (v_wmma_f32_16x16x32_bf16)
//  - GEMMs register-blocked: one wave = 64x32 output (4x2 WMMA subtiles)
//  - flash attention: 8-wave workgroup shares K/V chunks via LDS; chunks are
//    staged by the Tensor Data Mover (tensor_load_to_lds + s_wait_tensorcnt):
//    5-arg builtin on clang<23 (ROCm 7.2), 6-arg on clang>=23 (amdgpu-toolchain)
//  - global_prefetch_b8 overlaps the next panel/chunk
// Workspace ~47 MB.

typedef __bf16 bf16;
typedef __attribute__((ext_vector_type(16))) __bf16 v16bf;
typedef __attribute__((ext_vector_type(8)))  float  v8f;
typedef __attribute__((ext_vector_type(4)))  unsigned v4u;
typedef __attribute__((ext_vector_type(8)))  int      v8i;
typedef __attribute__((ext_vector_type(4)))  int      v4i;

#define DEV __device__ __forceinline__

#if defined(__AMDGCN__) && __has_builtin(__builtin_amdgcn_tensor_load_to_lds) && \
    __has_builtin(__builtin_amdgcn_s_wait_tensorcnt)
#define USE_TDM 1
#else
#define USE_TDM 0
#endif

#define XSTR2(x) #x
#define XSTR(x) XSTR2(x)
#pragma message("cdna5-probe: USE_TDM=" XSTR(USE_TDM) " clang=" XSTR(__clang_major__))

static constexpr int B  = 2;
static constexpr int L  = 2048;
static constexpr int MM = 2048;     // memory length
static constexpr int S  = 4096;     // MM + L
static constexpr int D  = 1024;
static constexpr int H  = 16;
static constexpr int DH = 64;
static constexpr int CH = 64;       // keys per shared chunk in attention

union Frag16 { unsigned u[8]; v16bf v; };

// A-operand fragment (16x32 bf16), ISA 7.12.2:
// half=lane>>4, row=lane&15; dword q holds K = (q>>2)*16 + half*8 + (q&3)*2 (+1)
DEV v16bf load_a_frag(const bf16* base, int stride, int lane) {
  const int half = (lane >> 4) & 1;
  const int row  = lane & 15;
  Frag16 f;
#pragma unroll
  for (int q = 0; q < 8; ++q) {
    const int k = ((q >> 2) << 4) + (half << 3) + ((q & 3) << 1);
    f.u[q] = *reinterpret_cast<const unsigned*>(base + (size_t)row * stride + k);
  }
  return f.v;
}

// B-operand fragment (32x16 bf16) from an n-major / k-contiguous buffer:
// half=lane>>4, col=lane&15; dword q holds K = (q>>2)*16 + half*8 + (q&3)*2 (+1)
DEV v16bf load_b_frag(const bf16* base, int stride, int lane) {
  const int half = (lane >> 4) & 1;
  const int col  = lane & 15;
  Frag16 f;
#pragma unroll
  for (int q = 0; q < 8; ++q) {
    const int k = ((q >> 2) << 4) + (half << 3) + ((q & 3) << 1);
    f.u[q] = *reinterpret_cast<const unsigned*>(base + (size_t)col * stride + k);
  }
  return f.v;
}

DEV v8f wmma_acc(v16bf a, v16bf b, v8f c) {
  return __builtin_amdgcn_wmma_f32_16x16x32_bf16(false, a, false, b, (short)0, c,
                                                 false, false);
}

DEV float redmax16(float x) {
  x = fmaxf(x, __shfl_xor(x, 1)); x = fmaxf(x, __shfl_xor(x, 2));
  x = fmaxf(x, __shfl_xor(x, 4)); x = fmaxf(x, __shfl_xor(x, 8));
  return x;
}
DEV float redsum16(float x) {
  x += __shfl_xor(x, 1); x += __shfl_xor(x, 2);
  x += __shfl_xor(x, 4); x += __shfl_xor(x, 8);
  return x;
}

#if USE_TDM
// Issue a TDM 2D tile load (global -> LDS). All arguments uniform.
// D# layout per CDNA5 ISA ch.8: group0 = {flags, lds_addr, global_addr, type=2},
// group1 = {data_size=2B, tensor dims, tile dims, dim0 stride}.
DEV void tdm_load_2d(unsigned lds_off, const void* gaddr,
                     unsigned tile_x, unsigned tile_y,
                     unsigned tensor_x, unsigned tensor_y,
                     unsigned long long stride_x) {
  const unsigned long long ga = (unsigned long long)(size_t)gaddr;
  v4u g0;
  g0[0] = 1u;                                   // count=1, user descriptor
  g0[1] = lds_off;                              // lds_addr (bytes)
  g0[2] = (unsigned)(ga & 0xffffffffu);         // global_addr[31:0]
  g0[3] = (unsigned)((ga >> 32) & 0x01ffffffu)  // global_addr[56:32]
          | (2u << 30);                         // type = 2 ("image")
  v8i g1;
  g1[0] = (int)(1u << 16);                      // data_size=1 -> 2 bytes; mask=0
  g1[1] = (int)((tensor_x & 0xffffu) << 16);    // tensor_dim0[15:0] @ bits 63:48
  g1[2] = (int)((tensor_x >> 16) | ((tensor_y & 0xffffu) << 16));
  g1[3] = (int)((tensor_y >> 16) | (tile_x << 16));   // tile_dim0 @ 127:112
  g1[4] = (int)(tile_y & 0xffffu);              // tile_dim1 (0 => 1D), tile_dim2=0
  g1[5] = (int)(unsigned)(stride_x & 0xffffffffu);    // tensor_dim0_stride[31:0]
  g1[6] = (int)(unsigned)((stride_x >> 32) & 0xffffu);
  g1[7] = 0;
  v4i g2 = {0, 0, 0, 0};
  v4i g3 = {0, 0, 0, 0};
#if __clang_major__ >= 23
  v8i g4 = {0, 0, 0, 0, 0, 0, 0, 0};
  __builtin_amdgcn_tensor_load_to_lds(g0, g1, g2, g3, g4, 0);
#else
  __builtin_amdgcn_tensor_load_to_lds(g0, g1, g2, g3, 0);
#endif
}
#endif

// ---------------- conversion kernels ----------------
__global__ void cvt_f32_bf16(const float* __restrict__ src, bf16* __restrict__ dst, int n) {
  for (int i = blockIdx.x * blockDim.x + threadIdx.x; i < n; i += gridDim.x * blockDim.x)
    dst[i] = (bf16)src[i];
}

__global__ void cvt_concat(const float* __restrict__ src, bf16* __restrict__ dst,
                           int perBatch, int extra, int off0, int n) {
  for (int i = blockIdx.x * blockDim.x + threadIdx.x; i < n; i += gridDim.x * blockDim.x)
    dst[(size_t)i + (size_t)(i / perBatch) * extra + off0] = (bf16)src[i];
}

__global__ void cvt_transpose(const float* __restrict__ src, bf16* __restrict__ dst,
                              int R, int C, int n) {
  for (int i = blockIdx.x * blockDim.x + threadIdx.x; i < n; i += gridDim.x * blockDim.x) {
    int r = i / C, c = i - r * C;
    dst[(size_t)c * R + r] = (bf16)src[i];
  }
}

// ---------------- register-blocked GEMM core ----------------
// One wave computes a 64(M) x 32(N) block: 4x2 subtiles, 8 WMMAs per K-step of 32.
DEV void gemm_block_4x2(const bf16* __restrict__ A, const bf16* __restrict__ Bt,
                        int K, int lane, v8f acc[4][2]) {
#pragma unroll
  for (int mi = 0; mi < 4; ++mi)
#pragma unroll
    for (int ni = 0; ni < 2; ++ni) acc[mi][ni] = (v8f){};
  for (int k0 = 0; k0 < K; k0 += 32) {
    __builtin_prefetch(A + k0 + 32, 0, 3);      // speculative: safe past the end
    __builtin_prefetch(Bt + k0 + 32, 0, 3);
    v16bf a[4], b[2];
#pragma unroll
    for (int mi = 0; mi < 4; ++mi)
      a[mi] = load_a_frag(A + (size_t)(mi * 16) * K + k0, K, lane);
#pragma unroll
    for (int ni = 0; ni < 2; ++ni)
      b[ni] = load_b_frag(Bt + (size_t)(ni * 16) * K + k0, K, lane);
#pragma unroll
    for (int mi = 0; mi < 4; ++mi)
#pragma unroll
      for (int ni = 0; ni < 2; ++ni)
        acc[mi][ni] = wmma_acc(a[mi], b[ni], acc[mi][ni]);
  }
}

// q projection:  A[B*L][D] x w_qT[D][D] -> qh head-major [B][H][L][DH]
__global__ void gemm_qh_kernel(const bf16* __restrict__ A, const bf16* __restrict__ Bt,
                               bf16* __restrict__ qh) {
  const int lane = threadIdx.x & 31, wave = threadIdx.x >> 5;
  const int gw = blockIdx.x * 8 + wave;
  const int nt = gw & 31, mt = gw >> 5;
  const int m0 = mt * 64, n0 = nt * 32;
  v8f acc[4][2];
  gemm_block_4x2(A + (size_t)m0 * D, Bt + (size_t)n0 * D, D, lane, acc);
  const int half = lane >> 4, lc = lane & 15;
#pragma unroll
  for (int mi = 0; mi < 4; ++mi)
#pragma unroll
    for (int ni = 0; ni < 2; ++ni) {
      const int col = n0 + ni * 16 + lc, h = col >> 6, d = col & 63;
#pragma unroll
      for (int r = 0; r < 8; ++r) {
        const int Mg = m0 + mi * 16 + r + 8 * half;
        const int b = Mg >> 11, i = Mg & (L - 1);
        qh[((((size_t)b * H + h) * L) + i) * DH + d] = (bf16)acc[mi][ni][r];
      }
    }
}

// kv projection: c[B*S][D] x w_kvT[128][D] -> K [B][S][DH], V^T [B][DH][S]
__global__ void gemm_kv_kernel(const bf16* __restrict__ A, const bf16* __restrict__ Bt,
                               bf16* __restrict__ kmat, bf16* __restrict__ vT) {
  const int lane = threadIdx.x & 31, wave = threadIdx.x >> 5;
  const int gw = blockIdx.x * 8 + wave;
  const int nt = gw & 3, mt = gw >> 2;
  const int m0 = mt * 64, n0 = nt * 32;
  v8f acc[4][2];
  gemm_block_4x2(A + (size_t)m0 * D, Bt + (size_t)n0 * D, D, lane, acc);
  const int half = lane >> 4, lc = lane & 15;
#pragma unroll
  for (int mi = 0; mi < 4; ++mi)
#pragma unroll
    for (int ni = 0; ni < 2; ++ni) {
      const int n = n0 + ni * 16 + lc;
#pragma unroll
      for (int r = 0; r < 8; ++r) {
        const int Mg = m0 + mi * 16 + r + 8 * half;
        const int b = Mg >> 12, j = Mg & (S - 1);
        if (n < DH)
          kmat[((size_t)b * S + j) * DH + n] = (bf16)acc[mi][ni][r];
        else
          vT[((size_t)b * DH + (n - DH)) * S + j] = (bf16)acc[mi][ni][r];
      }
    }
}

// output projection: attn[B*L][D] x w_cT[D][D] -> f32 out [B*L][D]
__global__ void gemm_out_kernel(const bf16* __restrict__ A, const bf16* __restrict__ Bt,
                                float* __restrict__ out) {
  const int lane = threadIdx.x & 31, wave = threadIdx.x >> 5;
  const int gw = blockIdx.x * 8 + wave;
  const int nt = gw & 31, mt = gw >> 5;
  const int m0 = mt * 64, n0 = nt * 32;
  v8f acc[4][2];
  gemm_block_4x2(A + (size_t)m0 * D, Bt + (size_t)n0 * D, D, lane, acc);
  const int half = lane >> 4, lc = lane & 15;
#pragma unroll
  for (int mi = 0; mi < 4; ++mi)
#pragma unroll
    for (int ni = 0; ni < 2; ++ni) {
      const int col = n0 + ni * 16 + lc;
#pragma unroll
      for (int r = 0; r < 8; ++r)
        out[(size_t)(m0 + mi * 16 + r + 8 * half) * D + col] = acc[mi][ni][r];
    }
}

// ---------------- fused flash attention ----------------
// Workgroup = 8 waves = 128 queries of one (b,h). K/V chunks (64 keys) staged in
// LDS once per workgroup (TDM when available) and consumed by all waves.
__global__ void attn_kernel(const bf16* __restrict__ qh, const bf16* __restrict__ kmat,
                            const bf16* __restrict__ vT, bf16* __restrict__ out) {
  __shared__ __align__(16) bf16 kls[CH * DH];       // K chunk  [j][d]   8 KB
  __shared__ __align__(16) bf16 vls[DH * CH];       // V^T chunk [d][j]  8 KB
  __shared__ __align__(16) bf16 plds[8][16 * CH];   // per-wave P stage 16 KB

  const int tid  = threadIdx.x;
  const int lane = tid & 31, wave = tid >> 5;
  const int gb = blockIdx.x;                        // B*H*(L/128) = 512 blocks
  const int ib = gb & 15, h = (gb >> 4) & 15, b = gb >> 8;
  const int it = ib * 8 + wave;
  const int half = lane >> 4, col = lane & 15;

  const bf16* qbase = qh + ((((size_t)b * H + h) * L) + (size_t)it * 16) * DH;
  const v16bf aq0 = load_a_frag(qbase, DH, lane);       // d = 0..31
  const v16bf aq1 = load_a_frag(qbase + 32, DH, lane);  // d = 32..63
  const bf16* kb = kmat + (size_t)b * S * DH;
  const bf16* vb = vT + (size_t)b * DH * S;
  bf16* pl = &plds[wave][0];

  float mrow[8], lrow[8];
  v8f o[4];
#pragma unroll
  for (int r = 0; r < 8; ++r) { mrow[r] = -3.0e38f; lrow[r] = 0.0f; }
#pragma unroll
  for (int f = 0; f < 4; ++f) o[f] = (v8f){};

  for (int j0 = 0; j0 < S; j0 += CH) {
    // ---- stage K (contiguous 8KB line) and V^T (64x64, stride S) into LDS ----
#if USE_TDM
    if (wave == 0) {
      tdm_load_2d((unsigned)(size_t)&kls[0], kb + (size_t)j0 * DH,
                  /*tile_x=*/CH * DH, /*tile_y=*/0,
                  /*tensor_x=*/CH * DH, /*tensor_y=*/1, /*stride_x=*/CH * DH);
      tdm_load_2d((unsigned)(size_t)&vls[0], vb + j0,
                  /*tile_x=*/CH, /*tile_y=*/DH,
                  /*tensor_x=*/S, /*tensor_y=*/DH, /*stride_x=*/S);
      __builtin_amdgcn_s_wait_tensorcnt(0);
    }
#else
    {
      const uint4 kv4 = *reinterpret_cast<const uint4*>(kb + (size_t)j0 * DH + tid * 16);
      *reinterpret_cast<uint4*>(kls + tid * 16) = kv4;
      const int vr = tid >> 2, vs = (tid & 3) * 16;   // 4 threads per V row
      const uint4 vv4 = *reinterpret_cast<const uint4*>(vb + (size_t)vr * S + j0 + vs);
      *reinterpret_cast<uint4*>(vls + vr * CH + vs) = vv4;
    }
#endif
    __syncthreads();
    // prefetch next chunk (speculative; dropped past the end)
    __builtin_prefetch(kb + (size_t)(j0 + CH) * DH + tid * 16, 0, 3);
    __builtin_prefetch(vb + (size_t)(tid >> 2) * S + j0 + CH + (tid & 3) * 16, 0, 3);

    // ---- scores: 4 j-subtiles x (2 k-halves of DH) = 8 WMMAs ----
    v8f s[4];
#pragma unroll
    for (int su = 0; su < 4; ++su) {
      s[su] = (v8f){};
      s[su] = wmma_acc(aq0, load_b_frag(kls + su * 16 * DH, DH, lane), s[su]);
      s[su] = wmma_acc(aq1, load_b_frag(kls + su * 16 * DH + 32, DH, lane), s[su]);
    }

    // ---- online softmax over the 64-key chunk ----
    float alpha[8];
#pragma unroll
    for (int r = 0; r < 8; ++r) {
      float e[4], cmax = -3.0e38f;
#pragma unroll
      for (int su = 0; su < 4; ++su) { e[su] = s[su][r] * 0.125f; cmax = fmaxf(cmax, e[su]); }
      cmax = redmax16(cmax);
      const float mn = fmaxf(mrow[r], cmax);
      const float al = __expf(mrow[r] - mn);
      float rs = 0.0f;
      const int row = r + 8 * half;
#pragma unroll
      for (int su = 0; su < 4; ++su) {
        const float p = __expf(e[su] - mn);
        rs += p;
        pl[row * CH + su * 16 + col] = (bf16)p;
      }
      lrow[r] = lrow[r] * al + redsum16(rs);
      mrow[r] = mn; alpha[r] = al;
    }
#pragma unroll
    for (int f = 0; f < 4; ++f)
#pragma unroll
      for (int r = 0; r < 8; ++r) o[f][r] *= alpha[r];

    // ---- O += P @ V : 4 d-subtiles x 2 j-halves = 8 WMMAs ----
    const v16bf ap0 = load_a_frag(pl, CH, lane);        // j = 0..31
    const v16bf ap1 = load_a_frag(pl + 32, CH, lane);   // j = 32..63
#pragma unroll
    for (int f = 0; f < 4; ++f) {
      o[f] = wmma_acc(ap0, load_b_frag(vls + f * 16 * CH, CH, lane), o[f]);
      o[f] = wmma_acc(ap1, load_b_frag(vls + f * 16 * CH + 32, CH, lane), o[f]);
    }
    __syncthreads();   // protect kls/vls before next chunk's stores
  }

  // normalize, store pre-transposed to (B, L, D) for the final GEMM
#pragma unroll
  for (int r = 0; r < 8; ++r) {
    const float inv = 1.0f / lrow[r];
    const size_t row = (size_t)b * L + (size_t)it * 16 + r + 8 * half;
#pragma unroll
    for (int f = 0; f < 4; ++f)
      out[row * D + h * DH + f * 16 + col] = (bf16)(o[f][r] * inv);
  }
}

// ---------------- host launch ----------------
extern "C" void kernel_launch(void* const* d_in, const int* in_sizes, int n_in,
                              void* d_out, int out_size, void* d_ws, size_t ws_size,
                              hipStream_t stream) {
  const float* q    = (const float*)d_in[0];
  const float* kv   = (const float*)d_in[1];
  const float* mem  = (const float*)d_in[2];
  const float* w_q  = (const float*)d_in[3];
  const float* w_kv = (const float*)d_in[4];
  const float* w_c  = (const float*)d_in[5];
  float* outp = (float*)d_out;

  size_t off = 0;
  auto alloc = [&](size_t elems) {
    bf16* p = (bf16*)((char*)d_ws + off);
    off += elems * sizeof(bf16);
    return p;
  };
  bf16* c_b  = alloc((size_t)B * S * D);
  bf16* q_b  = alloc((size_t)B * L * D);
  bf16* wqT  = alloc((size_t)D * D);
  bf16* wkvT = alloc((size_t)128 * D);
  bf16* wcT  = alloc((size_t)D * D);
  bf16* qh   = alloc((size_t)B * H * L * DH);
  bf16* kmat = alloc((size_t)B * S * DH);
  bf16* vTm  = alloc((size_t)B * DH * S);
  bf16* attn = alloc((size_t)B * L * D);
  (void)ws_size; (void)in_sizes; (void)n_in; (void)out_size;

  const int T = 256, GB = 4096;
  cvt_f32_bf16<<<GB, T, 0, stream>>>(q, q_b, B * L * D);
  cvt_concat<<<GB, T, 0, stream>>>(mem, c_b, MM * D, L * D, 0, B * MM * D);
  cvt_concat<<<GB, T, 0, stream>>>(kv, c_b, L * D, MM * D, MM * D, B * L * D);
  cvt_transpose<<<GB, T, 0, stream>>>(w_q, wqT, D, D, D * D);
  cvt_transpose<<<GB, T, 0, stream>>>(w_kv, wkvT, D, 128, D * 128);
  cvt_transpose<<<GB, T, 0, stream>>>(w_c, wcT, D, D, D * D);
  gemm_qh_kernel<<<256, T, 0, stream>>>(q_b, wqT, qh);        // 2048 waves
  gemm_kv_kernel<<<64, T, 0, stream>>>(c_b, wkvT, kmat, vTm); // 512 waves
  attn_kernel<<<512, T, 0, stream>>>(qh, kmat, vTm, attn);    // 512 WGs x 128 queries
  gemm_out_kernel<<<256, T, 0, stream>>>(attn, wcT, outp);
}
